// AnchorMixtureRNN_40656160424376
// MI455X (gfx1250) — compile-verified, matrix operand order checked by reading, not run on hardware
//
#include <hip/hip_runtime.h>

typedef __attribute__((ext_vector_type(16))) _Float16 v16h;
typedef __attribute__((ext_vector_type(8)))  float    v8f;

#define EMB_SCALE 22.62741699796952f
#define INV_SCALE 0.044194173824159216f

// ---------------------------------------------------------------- helpers

__device__ __forceinline__ int kmap(int k0, int e, int hf) {
    // 16-bit A/B fragment K index: lanes 0-15 -> K {0..7,16..23}, lanes 16-31 -> +8
    return k0 + (e & 7) + ((e >> 3) << 4) + (hf << 3);
}

__device__ __forceinline__ float wave_red_sum(float s) {
#pragma unroll
    for (int o = 16; o > 0; o >>= 1) s += __shfl_xor(s, o, 32);
    return s;
}

// CDNA5 async bulk copy global -> LDS (ASYNCcnt tracked), 16B granules.
// ldst/gsrc must be 16B aligned, bytes a multiple of 16*nthr is not required,
// but each issued chunk is 16B. Follows with s_wait_asynccnt 0.
__device__ __forceinline__ void async_copy_g2l(void* ldst, const void* gsrc, int bytes,
                                               int tid, int nthr) {
    unsigned base = (unsigned)(unsigned long long)ldst;   // low 32 bits = LDS offset
    const char* g = (const char*)gsrc;
    for (int off = tid * 16; off < bytes; off += nthr * 16) {
        unsigned l = base + (unsigned)off;
        const void* ga = g + off;
        asm volatile("global_load_async_to_lds_b128 %0, %1, off"
                     :: "v"(l), "v"(ga) : "memory");
    }
    asm volatile("s_wait_asynccnt 0x0" ::: "memory");
}

// block-wide mean / rstd over 512 threads (one value per thread)
__device__ __forceinline__ void block_mean_rstd(float u, float* red, int tid, int lane,
                                                int w, float& mean, float& rstd) {
    float s1 = u, s2 = u * u;
#pragma unroll
    for (int o = 16; o > 0; o >>= 1) { s1 += __shfl_xor(s1, o, 32); s2 += __shfl_xor(s2, o, 32); }
    if (lane == 0) { red[w] = s1; red[16 + w] = s2; }
    __syncthreads();
    if (tid == 0) {
        float a = 0.f, b = 0.f;
        for (int i = 0; i < 16; ++i) { a += red[i]; b += red[16 + i]; }
        red[32] = a; red[33] = b;
    }
    __syncthreads();
    mean = red[32] * (1.0f / 512.0f);
    float var = red[33] * (1.0f / 512.0f) - mean * mean;
    rstd = rsqrtf(var + 1e-6f);
    __syncthreads();
}

// device-scope generation barrier across all blocks of the grid
__device__ __forceinline__ void gbarrier(unsigned* cnt, unsigned* gen, unsigned nb, unsigned& lg) {
    __syncthreads();
    if (threadIdx.x == 0) {
        __threadfence();
        unsigned g = lg;
        if (atomicAdd(cnt, 1u) == nb - 1u) {
            atomicExch(cnt, 0u);
            __threadfence();
            atomicAdd(gen, 1u);
        } else {
            while (__hip_atomic_load(gen, __ATOMIC_ACQUIRE, __HIP_MEMORY_SCOPE_AGENT) == g)
                __builtin_amdgcn_s_sleep(2);
        }
        lg = g + 1u;
        __threadfence();
    }
    __syncthreads();
}

// ---------------------------------------------------------------- init

__global__ void k_init(unsigned* c) { if (threadIdx.x < 8) c[threadIdx.x] = 0u; }

// ---------------------------------------------------------------- generic WMMA GEMM
// C[M,N] = A[M,K] @ op(B) + bias ; bt=1: B is [N,K] (B^T product), bt=0: B is [K,N]
// f32 inputs, f16 WMMA compute, f32 and/or f16 outputs.
__global__ void k_gemm(const float* __restrict__ A, const float* __restrict__ Bm,
                       const float* __restrict__ bias, float* Cf, _Float16* Ch,
                       int M, int N, int K, int bt) {
    int lane = threadIdx.x & 31;
    int wid  = (blockIdx.x * (blockDim.x >> 5)) + (threadIdx.x >> 5);
    int nwav = gridDim.x * (blockDim.x >> 5);
    int mt = M >> 4, nt = N >> 4;
    int r = lane & 15, hf = lane >> 4;
    for (int tt = wid; tt < mt * nt; tt += nwav) {
        int m0 = (tt / nt) << 4, n0 = (tt % nt) << 4;
        v8f acc = {};
        for (int k0 = 0; k0 < K; k0 += 32) {
            v16h af, bf;
            const float* ap = A + (size_t)(m0 + r) * K;
#pragma unroll
            for (int e = 0; e < 16; ++e) af[e] = (_Float16)ap[kmap(k0, e, hf)];
            if (bt) {
                const float* bp = Bm + (size_t)(n0 + r) * K;
#pragma unroll
                for (int e = 0; e < 16; ++e) bf[e] = (_Float16)bp[kmap(k0, e, hf)];
            } else {
#pragma unroll
                for (int e = 0; e < 16; ++e)
                    bf[e] = (_Float16)Bm[(size_t)kmap(k0, e, hf) * N + (n0 + r)];
            }
            acc = __builtin_amdgcn_wmma_f32_16x16x32_f16(false, af, false, bf,
                                                         (short)0, acc, false, false);
        }
        int n = n0 + r;
        float bb = bias ? bias[n] : 0.0f;
#pragma unroll
        for (int q = 0; q < 8; ++q) {
            int m = m0 + q + (hf << 3);
            float v = acc[q] + bb;
            if (Cf) Cf[(size_t)m * N + n] = v;
            if (Ch) Ch[(size_t)m * N + n] = (_Float16)v;
        }
    }
}

__global__ void k_cgate(const float* __restrict__ aq, const float* __restrict__ bk, float* cg) {
    int m = threadIdx.x;
    if (m < 64) {
        float s = 0.f;
        for (int e = 0; e < 512; ++e) s += aq[m * 512 + e] * bk[e];
        cg[m] = s;
    }
}

// ---------------------------------------------------------------- encoder scan (persistent)
// grid = 8 GEMM blocks + 64 per-batch state blocks, block = 512 threads (16 waves)
__global__ void __launch_bounds__(512, 1)
k_enc_scan(const int* __restrict__ iseq, const float* __restrict__ emb_in,
           const float* __restrict__ Wq, const float* __restrict__ bq,
           const float* __restrict__ Wk, const float* __restrict__ bk,
           const float* __restrict__ Wv, const float* __restrict__ bv,
           const _Float16* __restrict__ mk, const _Float16* __restrict__ mv,
           const _Float16* __restrict__ wg, const float* __restrict__ cg,
           const float* __restrict__ ln1g, const float* __restrict__ ln1b,
           _Float16* zsc, float* qkv, float* av_out, unsigned* cnt, unsigned* gen) {
    extern __shared__ char smem[];
    __shared__ unsigned lg;
    if (threadIdx.x == 0) lg = 0u;
    const int tid = threadIdx.x, lane = tid & 31, w = tid >> 5;
    const unsigned NB = gridDim.x;

    if (blockIdx.x < 8) {
        // ---------------- GEMM role: 192-column slice of [Wq|Wk|Wv] in LDS ----------------
        const int a = blockIdx.x;
        _Float16* wsl = (_Float16*)smem;                      // [192][512] f16
        _Float16* zl  = (_Float16*)(smem + 192 * 512 * 2);    // [64][512]  f16
        float*    bl  = (float*)(smem + 192 * 512 * 2 + 64 * 512 * 2);
        for (int idx = tid; idx < 192 * 512; idx += 512) {
            int cl = idx >> 9, k = idx & 511;
            int cgl = a * 192 + cl;
            const float* Ws = (cgl < 512) ? Wq : (cgl < 1024 ? Wk : Wv);
            wsl[cl * 512 + k] = (_Float16)Ws[(size_t)(cgl & 511) * 512 + k];
        }
        if (tid < 192) {
            int cgl = a * 192 + tid;
            const float* bs = (cgl < 512) ? bq : (cgl < 1024 ? bk : bv);
            bl[tid] = bs[cgl & 511];
        }
        int r = lane & 15, hf = lane >> 4;
        // wave -> 1 m-tile, 3 n-tiles (A fragment reused across 3 WMMAs)
        const int mT = w >> 2, nT0 = (w & 3) * 3;
        const int m0 = mT << 4;
        for (int t = 0; t < 1024; ++t) {
            gbarrier(cnt, gen, NB, lg);                       // z(t) published
            async_copy_g2l(zl, zsc, 64 * 512 * 2, tid, 512);  // 64KB state broadcast
            __syncthreads();
            v8f acc0 = {}, acc1 = {}, acc2 = {};
            for (int k0 = 0; k0 < 512; k0 += 32) {
                v16h af, bf0, bf1, bf2;
                const _Float16* ap = zl + (m0 + r) * 512;
                const _Float16* bp0 = wsl + (((nT0 + 0) << 4) + r) * 512;
                const _Float16* bp1 = wsl + (((nT0 + 1) << 4) + r) * 512;
                const _Float16* bp2 = wsl + (((nT0 + 2) << 4) + r) * 512;
#pragma unroll
                for (int e = 0; e < 16; ++e) {
                    int k = kmap(k0, e, hf);
                    af[e] = ap[k]; bf0[e] = bp0[k]; bf1[e] = bp1[k]; bf2[e] = bp2[k];
                }
                acc0 = __builtin_amdgcn_wmma_f32_16x16x32_f16(false, af, false, bf0,
                                                              (short)0, acc0, false, false);
                acc1 = __builtin_amdgcn_wmma_f32_16x16x32_f16(false, af, false, bf1,
                                                              (short)0, acc1, false, false);
                acc2 = __builtin_amdgcn_wmma_f32_16x16x32_f16(false, af, false, bf2,
                                                              (short)0, acc2, false, false);
            }
#pragma unroll
            for (int j = 0; j < 3; ++j) {
                v8f acc = (j == 0) ? acc0 : (j == 1 ? acc1 : acc2);
                int nl = ((nT0 + j) << 4) + r;
                float bb = bl[nl];
#pragma unroll
                for (int q = 0; q < 8; ++q) {
                    int m = m0 + q + (hf << 3);
                    qkv[m * 1536 + a * 192 + nl] = acc[q] + bb;
                }
            }
            gbarrier(cnt, gen, NB, lg);                       // qkv(t) ready
        }
    } else {
        // ---------------- state role: one batch row, av + mover_k/v + W_gate in LDS ----------------
        const int b = blockIdx.x - 8;
        _Float16* avl = (_Float16*)smem;                      // [64][512]
        _Float16* mkl = avl + 64 * 512;
        _Float16* mvl = mkl + 64 * 512;
        _Float16* wgl = mvl + 64 * 512;
        float* fl  = (float*)(wgl + 64 * 512);
        float* g1  = fl;          float* b1  = fl + 512;
        float* qzL = fl + 1024;   float* kzL = fl + 1536;  float* vzL = fl + 2048;
        float* znL = fl + 2560;
        float* sL  = fl + 3072;   // 65 (+pad)
        float* gtL = fl + 3152;   // 64
        float* red = fl + 3216;   // 48
        float* cgl = fl + 3264;   // 64
        async_copy_g2l(mkl, mk, 64 * 512 * 2, tid, 512);
        async_copy_g2l(mvl, mv, 64 * 512 * 2, tid, 512);
        async_copy_g2l(wgl, wg, 64 * 512 * 2, tid, 512);
        for (int idx = tid; idx < 64 * 512; idx += 512) avl[idx] = (_Float16)0.0f;
        g1[tid] = ln1g[tid]; b1[tid] = ln1b[tid];
        if (tid < 64) cgl[tid] = cg[tid];
        const int e = tid;
        float zprev = 0.0f;
        for (int t = 0; t < 1024; ++t) {
            int tok = iseq[b * 1024 + t];
            float zpre = zprev + emb_in[(size_t)tok * 512 + e] * EMB_SCALE;
            zsc[b * 512 + e] = (_Float16)zpre;
            if (t + 1 < 1024) {                               // hide next gather under GEMM
                int tn = iseq[b * 1024 + t + 1];
                __builtin_prefetch(emb_in + (size_t)tn * 512 + e, 0, 1);
            }
            gbarrier(cnt, gen, NB, lg);                       // publish z
            gbarrier(cnt, gen, NB, lg);                       // wait qkv
            qzL[e] = qkv[b * 1536 + e];
            kzL[e] = qkv[b * 1536 + 512 + e];
            vzL[e] = qkv[b * 1536 + 1024 + e];
            __syncthreads();
            // scores: 64 mover keys + self key, wave-parallel dots
            for (int j = w; j < 65; j += 16) {
                float s = 0.f;
                if (j < 64) { for (int i2 = lane; i2 < 512; i2 += 32) s += qzL[i2] * (float)mkl[j * 512 + i2]; }
                else        { for (int i2 = lane; i2 < 512; i2 += 32) s += qzL[i2] * kzL[i2]; }
                s = wave_red_sum(s);
                if (lane == 0) sL[j] = s * INV_SCALE;
            }
            __syncthreads();
            if (tid == 0) {
                float mx = sL[0];
                for (int j = 1; j < 65; ++j) mx = fmaxf(mx, sL[j]);
                float sm = 0.f;
                for (int j = 0; j < 65; ++j) { float ev = __expf(sL[j] - mx); sL[j] = ev; sm += ev; }
                float inv = 1.0f / sm;
                for (int j = 0; j < 65; ++j) sL[j] *= inv;
            }
            __syncthreads();
            float u = zpre + sL[64] * vzL[e];
            for (int j = 0; j < 64; ++j) u += sL[j] * (float)mvl[j * 512 + e];
            float mean, rstd;
            block_mean_rstd(u, red, tid, lane, w, mean, rstd);
            float zn = g1[e] * (u - mean) * rstd + b1[e];
            znL[e] = zn;
            __syncthreads();
            // anchor gates via folded W_gate
            for (int m = w; m < 64; m += 16) {
                float s = 0.f;
                for (int i2 = lane; i2 < 512; i2 += 32) s += znL[i2] * (float)wgl[m * 512 + i2];
                s = wave_red_sum(s);
                if (lane == 0) gtL[m] = 1.0f / (1.0f + __expf(-((s + cgl[m]) * INV_SCALE)));
            }
            __syncthreads();
            // av <- LN((1-g)*av + g*z), one wave per 4 rows, row-LN via wave reduce
            for (int m = w * 4; m < w * 4 + 4; ++m) {
                float gv = gtL[m], vals[16], p1 = 0.f, p2 = 0.f;
#pragma unroll
                for (int c = 0; c < 16; ++c) {
                    int e2 = lane + (c << 5);
                    float uu = (1.0f - gv) * (float)avl[m * 512 + e2] + gv * znL[e2];
                    vals[c] = uu; p1 += uu; p2 += uu * uu;
                }
#pragma unroll
                for (int o = 16; o > 0; o >>= 1) { p1 += __shfl_xor(p1, o, 32); p2 += __shfl_xor(p2, o, 32); }
                float mn = p1 * (1.0f / 512.0f);
                float vr = p2 * (1.0f / 512.0f) - mn * mn;
                float rs = rsqrtf(vr + 1e-6f);
#pragma unroll
                for (int c = 0; c < 16; ++c) {
                    int e2 = lane + (c << 5);
                    avl[m * 512 + e2] = (_Float16)(g1[e2] * (vals[c] - mn) * rs + b1[e2]);
                }
            }
            __syncthreads();
            zprev = zn;
        }
        for (int m = 0; m < 64; ++m)
            av_out[((size_t)(b * 64 + m)) * 512 + e] = (float)avl[m * 512 + e];
    }
}

// ---------------------------------------------------------------- decoder scan (persistent)
__global__ void __launch_bounds__(512, 1)
k_dec_scan(const int* __restrict__ oseq, const float* __restrict__ emb_out,
           const float* __restrict__ Wqd, const float* __restrict__ bqd,
           const _Float16* __restrict__ kvk, const _Float16* __restrict__ kvv,
           const float* __restrict__ ln2g, const float* __restrict__ ln2b,
           _Float16* zsc, float* qsc, float* zdec_out, unsigned* cnt, unsigned* gen) {
    extern __shared__ char smem[];
    __shared__ unsigned lg;
    if (threadIdx.x == 0) lg = 0u;
    const int tid = threadIdx.x, lane = tid & 31, w = tid >> 5;
    const unsigned NB = gridDim.x;

    if (blockIdx.x < 8) {
        const int a = blockIdx.x;                             // 64-column slice of dec_Wq
        _Float16* wsl = (_Float16*)smem;                      // [64][512]
        _Float16* zl  = (_Float16*)(smem + 64 * 512 * 2);
        float*    bl  = (float*)(smem + 2 * 64 * 512 * 2);
        for (int idx = tid; idx < 64 * 512; idx += 512) {
            int cl = idx >> 9, k = idx & 511;
            wsl[cl * 512 + k] = (_Float16)Wqd[(size_t)(a * 64 + cl) * 512 + k];
        }
        if (tid < 64) bl[tid] = bqd[a * 64 + tid];
        int r = lane & 15, hf = lane >> 4;
        for (int t = 0; t < 1024; ++t) {
            gbarrier(cnt, gen, NB, lg);
            async_copy_g2l(zl, zsc, 64 * 512 * 2, tid, 512);
            __syncthreads();
            {
                int m0 = (w >> 2) << 4, n0 = (w & 3) << 4;    // 16 tiles / 16 waves
                v8f acc = {};
                for (int k0 = 0; k0 < 512; k0 += 32) {
                    v16h af, bf;
                    const _Float16* ap = zl + (m0 + r) * 512;
                    const _Float16* bp = wsl + (n0 + r) * 512;
#pragma unroll
                    for (int e = 0; e < 16; ++e) {
                        int k = kmap(k0, e, hf);
                        af[e] = ap[k]; bf[e] = bp[k];
                    }
                    acc = __builtin_amdgcn_wmma_f32_16x16x32_f16(false, af, false, bf,
                                                                 (short)0, acc, false, false);
                }
                int nl = n0 + r;
                float bb = bl[nl];
#pragma unroll
                for (int q = 0; q < 8; ++q) {
                    int m = m0 + q + (hf << 3);
                    qsc[m * 512 + a * 64 + nl] = acc[q] + bb;
                }
            }
            gbarrier(cnt, gen, NB, lg);
        }
    } else {
        const int b = blockIdx.x - 8;
        _Float16* kvkL = (_Float16*)smem;                     // [64][512]
        _Float16* kvvL = kvkL + 64 * 512;
        float* fl = (float*)(kvvL + 64 * 512);
        float* g2 = fl;        float* b2 = fl + 512;
        float* qL = fl + 1024;
        float* sL = fl + 1536; // 64 (+pad)
        float* red = fl + 1616;
        async_copy_g2l(kvkL, kvk + (size_t)b * 32768, 64 * 512 * 2, tid, 512);
        async_copy_g2l(kvvL, kvv + (size_t)b * 32768, 64 * 512 * 2, tid, 512);
        g2[tid] = ln2g[tid]; b2[tid] = ln2b[tid];
        const int e = tid;
        float zprev = 0.0f;
        for (int t = 0; t < 1024; ++t) {
            int tok = oseq[b * 1024 + t];
            float zpre = zprev + emb_out[(size_t)tok * 512 + e] * EMB_SCALE;
            zsc[b * 512 + e] = (_Float16)zpre;
            if (t + 1 < 1024) {
                int tn = oseq[b * 1024 + t + 1];
                __builtin_prefetch(emb_out + (size_t)tn * 512 + e, 0, 1);
            }
            gbarrier(cnt, gen, NB, lg);
            gbarrier(cnt, gen, NB, lg);
            qL[e] = qsc[b * 512 + e];
            __syncthreads();
            for (int m = w; m < 64; m += 16) {
                float s = 0.f;
                for (int i2 = lane; i2 < 512; i2 += 32) s += qL[i2] * (float)kvkL[m * 512 + i2];
                s = wave_red_sum(s);
                if (lane == 0) sL[m] = s * INV_SCALE;
            }
            __syncthreads();
            if (tid == 0) {
                float mx = sL[0];
                for (int m = 1; m < 64; ++m) mx = fmaxf(mx, sL[m]);
                float sm = 0.f;
                for (int m = 0; m < 64; ++m) { float ev = __expf(sL[m] - mx); sL[m] = ev; sm += ev; }
                float inv = 1.0f / sm;
                for (int m = 0; m < 64; ++m) sL[m] *= inv;
            }
            __syncthreads();
            float u = zpre;
            for (int m = 0; m < 64; ++m) u += sL[m] * (float)kvvL[m * 512 + e];
            float mean, rstd;
            block_mean_rstd(u, red, tid, lane, w, mean, rstd);
            zprev = g2[e] * (u - mean) * rstd + b2[e];
        }
        zdec_out[b * 512 + e] = zprev;
    }
}

// ---------------------------------------------------------------- log-softmax over V=32000
__global__ void k_logsoftmax(const float* __restrict__ logits, float* out) {
    __shared__ float red[18];
    int b = blockIdx.x, tid = threadIdx.x, lane = tid & 31, w = tid >> 5;
    const float* row = logits + (size_t)b * 32000;
    float mx = -3.402823466e38f;
    for (int i = tid; i < 32000; i += 256) mx = fmaxf(mx, row[i]);
#pragma unroll
    for (int o = 16; o > 0; o >>= 1) mx = fmaxf(mx, __shfl_xor(mx, o, 32));
    if (lane == 0) red[w] = mx;
    __syncthreads();
    if (tid == 0) { float m2 = red[0]; for (int i = 1; i < 8; ++i) m2 = fmaxf(m2, red[i]); red[16] = m2; }
    __syncthreads();
    float MX = red[16];
    float sm = 0.f;
    for (int i = tid; i < 32000; i += 256) sm += __expf(row[i] - MX);
    sm = wave_red_sum(sm);
    if (lane == 0) red[w] = sm;
    __syncthreads();
    if (tid == 0) { float s2 = 0.f; for (int i = 0; i < 8; ++i) s2 += red[i]; red[17] = logf(s2); }
    __syncthreads();
    float LS = MX + red[17];
    for (int i = tid; i < 32000; i += 256) out[(size_t)b * 32000 + i] = row[i] - LS;
}

// ---------------------------------------------------------------- host

extern "C" void kernel_launch(void* const* d_in, const int* in_sizes, int n_in,
                              void* d_out, int out_size, void* d_ws, size_t ws_size,
                              hipStream_t stream) {
    (void)in_sizes; (void)n_in; (void)out_size; (void)ws_size;
    const int*   iseq    = (const int*)  d_in[0];
    const int*   oseq    = (const int*)  d_in[1];
    const float* emb_in  = (const float*)d_in[2];
    const float* emb_out = (const float*)d_in[3];
    const float* vocW    = (const float*)d_in[4];
    const float* vocb    = (const float*)d_in[5];
    const float* movW    = (const float*)d_in[6];
    const float* anckW   = (const float*)d_in[7];
    const float* eWq = (const float*)d_in[8];  const float* ebq = (const float*)d_in[9];
    const float* eWk = (const float*)d_in[10]; const float* ebk = (const float*)d_in[11];
    const float* eWv = (const float*)d_in[12]; const float* ebv = (const float*)d_in[13];
    const float* aWq = (const float*)d_in[14]; const float* abq = (const float*)d_in[15];
    const float* aWk = (const float*)d_in[16]; const float* abk = (const float*)d_in[17];
    const float* dWq = (const float*)d_in[18]; const float* dbq = (const float*)d_in[19];
    const float* dWk = (const float*)d_in[20]; const float* dbk = (const float*)d_in[21];
    const float* dWv = (const float*)d_in[22]; const float* dbv = (const float*)d_in[23];
    const float* l1g = (const float*)d_in[24]; const float* l1b = (const float*)d_in[25];
    const float* l2g = (const float*)d_in[26]; const float* l2b = (const float*)d_in[27];

    char* ws = (char*)d_ws;
    unsigned* cnts  = (unsigned*)(ws + 0);
    _Float16* mk    = (_Float16*)(ws + 256);
    _Float16* mv    = (_Float16*)(ws + 65792);
    _Float16* wg    = (_Float16*)(ws + 131328);
    float*    cg    = (float*)   (ws + 196864);
    float*    aq    = (float*)   (ws + 197120);
    _Float16* zsc   = (_Float16*)(ws + 328192);
    float*    qkv   = (float*)   (ws + 393728);
    float*    qsc   = (float*)   (ws + 786944);
    _Float16* kvk   = (_Float16*)(ws + 918016);
    _Float16* kvv   = (_Float16*)(ws + 5112320);
    float*    logit = (float*)   (ws + 9306624);

    float* out_z  = (float*)d_out;          // [64,512]
    float* out_av = out_z + 32768;          // [64,64,512]
    float* out_lp = out_z + 2129920;        // [64,32000]

    const size_t SH_ENC = 275456;           // max of GEMM-role / state-role LDS footprints
    const size_t SH_DEC = 137728;

    k_init<<<1, 32, 0, stream>>>(cnts);
    // aq = anc_key_W @ anc_Wq^T + anc_bq
    k_gemm<<<16, 256, 0, stream>>>(anckW, aWq, abq, aq, (_Float16*)nullptr, 64, 512, 512, 1);
    k_cgate<<<1, 64, 0, stream>>>(aq, abk, cg);
    // W_gate = aq @ anc_Wk  (non-transposed B)
    k_gemm<<<16, 256, 0, stream>>>(aq, aWk, (const float*)nullptr, (float*)nullptr, wg, 64, 512, 512, 0);
    // mover_k / mover_v
    k_gemm<<<16, 256, 0, stream>>>(movW, eWk, ebk, (float*)nullptr, mk, 64, 512, 512, 1);
    k_gemm<<<16, 256, 0, stream>>>(movW, eWv, ebv, (float*)nullptr, mv, 64, 512, 512, 1);
    // fused persistent encoder scan (8 GEMM blocks + 64 state blocks)
    k_enc_scan<<<72, 512, SH_ENC, stream>>>(iseq, emb_in, eWq, ebq, eWk, ebk, eWv, ebv,
                                            mk, mv, wg, cg, l1g, l1b,
                                            zsc, qkv, out_av, cnts + 0, cnts + 1);
    // kv_k / kv_v = av @ dec_W{k,v}^T + b   ([4096,512] x [512,512])
    k_gemm<<<1024, 256, 0, stream>>>(out_av, dWk, dbk, (float*)nullptr, kvk, 4096, 512, 512, 1);
    k_gemm<<<1024, 256, 0, stream>>>(out_av, dWv, dbv, (float*)nullptr, kvv, 4096, 512, 512, 1);
    // fused persistent decoder scan
    k_dec_scan<<<72, 512, SH_DEC, stream>>>(oseq, emb_out, dWq, dbq, kvk, kvv, l2g, l2b,
                                            zsc, qsc, out_z, cnts + 2, cnts + 3);
    // vocab projection + log-softmax
    k_gemm<<<1000, 256, 0, stream>>>(out_z, vocW, vocb, logit, (_Float16*)nullptr, 64, 32000, 512, 1);
    k_logsoftmax<<<64, 256, 0, stream>>>(logit, out_lp);
}
